// LSTM_50955491999944
// MI455X (gfx1250) — matrix operand order repaired and checked by reading, not compiled, not run on hardware
//
#include <hip/hip_runtime.h>
#include <math.h>

typedef __attribute__((ext_vector_type(16))) __bf16 v16bf;
typedef __attribute__((ext_vector_type(8)))  __bf16 v8bf;
typedef __attribute__((ext_vector_type(8)))  float  v8f;
typedef __attribute__((ext_vector_type(4)))  float  v4f;

#define T_STEPS 512
#define BATCH   64
#define IDIM    512
#define HDIM    512
#define MROWS   (T_STEPS * BATCH)   // 32768 rows for the input projection GEMM

// ---------------------------------------------------------------------------
// Fast activations (gfx1250 TRANS: v_exp_f32 + v_rcp_f32)
// ---------------------------------------------------------------------------
static __device__ __forceinline__ float sigmoid_f(float x) {
    return __builtin_amdgcn_rcpf(1.0f + __expf(-x));
}
static __device__ __forceinline__ float tanh_f(float x) {
    // overflow-safe: e in (0,1]
    float e = __expf(-2.0f * fabsf(x));
    float t = (1.0f - e) * __builtin_amdgcn_rcpf(1.0f + e);
    return copysignf(t, x);
}

static __device__ __forceinline__ v8f wmma_bf16(v16bf a, v16bf b, v8f c) {
    // D = A(16x32 bf16) * B(32x16 bf16) + C(16x16 f32)
    return __builtin_amdgcn_wmma_f32_16x16x32_bf16(
        /*neg_a=*/false, a, /*neg_b=*/false, b,
        /*c_mod=*/(short)0, c, /*reuse_a=*/false, /*reuse_b=*/false);
}

// ---------------------------------------------------------------------------
// Kernel 1: transpose + convert all 8 weight matrices f32[K][N] -> bf16[N][K]
// mats 0..3 = Wx gates (a,i,f,o), mats 4..7 = Wh gates
// ---------------------------------------------------------------------------
__global__ __launch_bounds__(256) void prep_weights(
    const float* __restrict__ wa_x, const float* __restrict__ wi_x,
    const float* __restrict__ wf_x, const float* __restrict__ wo_x,
    const float* __restrict__ wa_h, const float* __restrict__ wi_h,
    const float* __restrict__ wf_h, const float* __restrict__ wo_h,
    __bf16* __restrict__ WxT, __bf16* __restrict__ WhT)
{
    int idx = blockIdx.x * 256 + threadIdx.x;        // 0 .. 8*512*512-1
    int mat = idx >> 18;                             // 512*512 = 2^18
    int r   = idx & 262143;
    int n   = r >> 9;
    int k   = r & 511;
    const float* src =
        (mat == 0) ? wa_x : (mat == 1) ? wi_x : (mat == 2) ? wf_x :
        (mat == 3) ? wo_x : (mat == 4) ? wa_h : (mat == 5) ? wi_h :
        (mat == 6) ? wf_h : wo_h;
    __bf16* dst = ((mat < 4) ? WxT : WhT) + (size_t)(mat & 3) * (512 * 512);
    dst[(size_t)n * 512 + k] = (__bf16)src[(size_t)k * 512 + n];
}

// ---------------------------------------------------------------------------
// Kernel 2: zero h (both parity buffers, bf16) and s (f32)
// ---------------------------------------------------------------------------
__global__ __launch_bounds__(256) void init_state(__bf16* __restrict__ hbuf,
                                                  float* __restrict__ sbuf)
{
    int i = blockIdx.x * 256 + threadIdx.x;          // 65536 threads
    hbuf[i] = (__bf16)0.0f;                          // 2 * 64*512 bf16
    if (i < BATCH * HDIM) sbuf[i] = 0.0f;
}

// ---------------------------------------------------------------------------
// Kernel 3: input projections Xg[g][m][n] = X[m] @ Wx_g + b_g  (m = t*B + b)
// One wave computes a 16x64 strip (4 accumulator tiles, A reused 4x).
// Software-pipelined K loop (prefetch A f32 + 4 B fragments one K-step ahead),
// LDS-staged coalesced b128 stores of the strip.
// ---------------------------------------------------------------------------
__global__ __launch_bounds__(256) void xproj_kernel(
    const float* __restrict__ X, const __bf16* __restrict__ WxT,
    const float* __restrict__ ba, const float* __restrict__ bi,
    const float* __restrict__ bfv, const float* __restrict__ bo,
    float* __restrict__ Xg)
{
    __shared__ float stile[8][16][68];   // 68-pad: bank-conflict-free + 16B aligned

    const int lane = threadIdx.x & 31;
    const int widx = threadIdx.x >> 5;
    const int gw   = blockIdx.x * 8 + widx;                 // global wave id
    const int g    = gw >> 14;                              // 2048 mtiles * 8 groups
    const int rem  = gw & 16383;
    const int mt   = rem >> 3;                              // M tile (0..2047)
    const int ng   = rem & 7;                               // 64-wide N group (0..7)

    const int rlane = lane & 15;
    const int half  = lane >> 4;     // 0: lanes 0-15, 1: lanes 16-31
    const int kb    = half * 16;     // A: K base inside 32-chunk
    const int ks    = half * 16;     // B: K base inside 32-chunk

    const int row  = mt * 16 + rlane;
    const int colb = ng * 64;
    const __bf16* wg   = WxT + (size_t)g * (HDIM * IDIM);
    const float*  xrow = X + (size_t)row * IDIM + kb;
    const __bf16* bcol = wg + (size_t)(colb + rlane) * IDIM + ks;

    v8f acc[4];
#pragma unroll
    for (int s = 0; s < 4; ++s) acc[s] = (v8f)0.0f;

    // --- software pipeline: double-buffered A (raw f32) and B fragments ---
    v4f   fA[2][4];
    v16bf bb[2][4];
    {
        const float* p = xrow;
        fA[0][0] = *(const v4f*)(p);
        fA[0][1] = *(const v4f*)(p + 4);
        fA[0][2] = *(const v4f*)(p + 8);
        fA[0][3] = *(const v4f*)(p + 12);
#pragma unroll
        for (int s = 0; s < 4; ++s)
            bb[0][s] = *(const v16bf*)(bcol + (size_t)s * 16 * IDIM);
    }

#pragma unroll 2
    for (int k0 = 0; k0 < IDIM; k0 += 32) {
        const int cur = (k0 >> 5) & 1;   // compile-time const per unrolled body
        const int nxt = cur ^ 1;
        if (k0 + 32 < IDIM) {            // prefetch next K-step
            const float* p = xrow + k0 + 32;
            fA[nxt][0] = *(const v4f*)(p);
            fA[nxt][1] = *(const v4f*)(p + 4);
            fA[nxt][2] = *(const v4f*)(p + 8);
            fA[nxt][3] = *(const v4f*)(p + 12);
#pragma unroll
            for (int s = 0; s < 4; ++s)
                bb[nxt][s] = *(const v16bf*)(bcol + (size_t)s * 16 * IDIM + (k0 + 32));
        }
        // convert current A fragment f32 -> bf16
        v16bf a;
#pragma unroll
        for (int e = 0; e < 4; ++e) {
            a[e]      = (__bf16)fA[cur][0][e];
            a[4 + e]  = (__bf16)fA[cur][1][e];
            a[8 + e]  = (__bf16)fA[cur][2][e];
            a[12 + e] = (__bf16)fA[cur][3][e];
        }
#pragma unroll
        for (int s = 0; s < 4; ++s)
            acc[s] = wmma_bf16(a, bb[cur][s], acc[s]);
    }

    // --- bias + stage C tiles in LDS (WMMA C layout -> row-major) ---
    const float* bias = (g == 0) ? ba : (g == 1) ? bi : (g == 2) ? bfv : bo;
#pragma unroll
    for (int s = 0; s < 4; ++s) {
        float bv = bias[colb + s * 16 + rlane];
#pragma unroll
        for (int v = 0; v < 8; ++v)
            stile[widx][half * 8 + v][s * 16 + rlane] = acc[s][v] + bv;
    }
    __builtin_amdgcn_wave_barrier();     // same-wave LDS RAW: DS ops in order

    // --- coalesced b128 stores: 16 lanes cover one 256B row segment ---
    float* outg = Xg + (size_t)g * MROWS * HDIM + (size_t)(mt * 16) * HDIM + colb;
#pragma unroll
    for (int it = 0; it < 8; ++it) {
        int flat = it * 32 + lane;       // 0..255 vec4s of the 16x64 strip
        int r    = flat >> 4;
        int c4   = (flat & 15) * 4;
        v4f vvv = *(const v4f*)&stile[widx][r][c4];
        *(v4f*)(outg + (size_t)r * HDIM + c4) = vvv;
    }
}

// ---------------------------------------------------------------------------
// Kernel 4: one recurrent step.
// 128 workgroups (4 M-tiles x 32 N-tiles), 4 waves each (one per gate).
// Wave: 16x16 tile of h_{t-1} @ Wg_h via 16 WMMAs (double-buffered K loop)
// -> LDS; then all 128 threads combine the 4 gates.
// ---------------------------------------------------------------------------
__global__ __launch_bounds__(128) void lstm_step_kernel(
    const __bf16* __restrict__ hprev, __bf16* __restrict__ hnext,
    const __bf16* __restrict__ WhT, const float* __restrict__ Xg,
    float* __restrict__ sbuf, float* __restrict__ out, int t)
{
    __shared__ float lds[4][16][16];

    const int lane = threadIdx.x & 31;
    const int g    = threadIdx.x >> 5;        // gate = wave id
    const int mt   = blockIdx.x & 3;
    const int nt   = blockIdx.x >> 2;

    const int rlane = lane & 15;
    const int half  = lane >> 4;
    const int kb    = half * 16;
    const int ks    = half * 16;

    const int row = mt * 16 + rlane;
    const int col = nt * 16 + rlane;
    const __bf16* wcol = WhT + (size_t)g * (HDIM * HDIM) + (size_t)col * HDIM + ks;
    const __bf16* hrow = hprev + (size_t)row * HDIM + kb;

    // software-pipelined K loop
    v8bf  alo[2], ahi[2];
    v16bf bB[2];
    alo[0] = *(const v8bf*)(hrow);
    ahi[0] = *(const v8bf*)(hrow + 8);
    bB[0]  = *(const v16bf*)(wcol);

    v8f acc = (v8f)0.0f;
#pragma unroll 2
    for (int k0 = 0; k0 < HDIM; k0 += 32) {
        const int cur = (k0 >> 5) & 1;
        const int nxt = cur ^ 1;
        if (k0 + 32 < HDIM) {
            alo[nxt] = *(const v8bf*)(hrow + k0 + 32);
            ahi[nxt] = *(const v8bf*)(hrow + k0 + 40);
            bB[nxt]  = *(const v16bf*)(wcol + k0 + 32);
        }
        v16bf a = __builtin_shufflevector(alo[cur], ahi[cur],
            0, 1, 2, 3, 4, 5, 6, 7, 8, 9, 10, 11, 12, 13, 14, 15);
        acc = wmma_bf16(a, bB[cur], acc);
    }

    const int m0 = half * 8;
#pragma unroll
    for (int v = 0; v < 8; ++v) lds[g][m0 + v][rlane] = acc[v];
    __syncthreads();

    // elementwise combine: 128 threads x 2 elements = 256 tile elements
#pragma unroll
    for (int j = 0; j < 2; ++j) {
        int e = threadIdx.x + j * 128;
        int m = e >> 4, n = e & 15;
        int brow = mt * 16 + m;
        int hcol = nt * 16 + n;
        size_t gidx = ((size_t)t * BATCH + brow) * HDIM + hcol;   // index into [T*B, H]
        size_t sidx = (size_t)brow * HDIM + hcol;

        const size_t GST = (size_t)MROWS * HDIM;
        float xa = Xg[0 * GST + gidx];
        float xi = Xg[1 * GST + gidx];
        float xf = Xg[2 * GST + gidx];
        float xo = Xg[3 * GST + gidx];

        float av = tanh_f  (xa + lds[0][m][n]);
        float iv = sigmoid_f(xi + lds[1][m][n]);
        float fv = sigmoid_f(xf + lds[2][m][n]);
        float ov = sigmoid_f(xo + lds[3][m][n]);

        float sn = av * iv + sbuf[sidx] * fv;
        float hn = tanh_f(sn) * ov;

        sbuf[sidx]  = sn;
        out[gidx]   = hn;                 // [T,B,H] f32 output
        hnext[sidx] = (__bf16)hn;         // bf16 h for next step's GEMM
    }
}

// ---------------------------------------------------------------------------
// Host-side orchestration
// Inputs: X, Wa_x, Wi_x, Wf_x, Wo_x, Wa_h, Wi_h, Wf_h, Wo_h, ba, bi, bf, bo
// ---------------------------------------------------------------------------
extern "C" void kernel_launch(void* const* d_in, const int* in_sizes, int n_in,
                              void* d_out, int out_size, void* d_ws, size_t ws_size,
                              hipStream_t stream)
{
    (void)in_sizes; (void)n_in; (void)out_size; (void)ws_size;

    const float* X    = (const float*)d_in[0];
    const float* Wa_x = (const float*)d_in[1];
    const float* Wi_x = (const float*)d_in[2];
    const float* Wf_x = (const float*)d_in[3];
    const float* Wo_x = (const float*)d_in[4];
    const float* Wa_h = (const float*)d_in[5];
    const float* Wi_h = (const float*)d_in[6];
    const float* Wf_h = (const float*)d_in[7];
    const float* Wo_h = (const float*)d_in[8];
    const float* ba   = (const float*)d_in[9];
    const float* bi   = (const float*)d_in[10];
    const float* bfp  = (const float*)d_in[11];
    const float* bo   = (const float*)d_in[12];

    // Workspace carve-up (all offsets are large power-of-two aligned):
    //   Xg  : 4 * 32768 * 512 f32  = 256 MB
    //   WxT : 4 * 512*512 bf16    =   2 MB
    //   WhT : 4 * 512*512 bf16    =   2 MB
    //   hbuf: 2 * 64*512 bf16     = 128 KB (ping-pong)
    //   sbuf: 64*512 f32          = 128 KB
    char*   ws   = (char*)d_ws;
    float*  Xg   = (float*)ws;
    __bf16* WxT  = (__bf16*)(ws + (size_t)4 * MROWS * HDIM * sizeof(float));
    __bf16* WhT  = WxT + (size_t)4 * HDIM * IDIM;
    __bf16* hbuf = WhT + (size_t)4 * HDIM * HDIM;
    float*  sbuf = (float*)(hbuf + (size_t)2 * BATCH * HDIM);

    // 1) transpose + convert weights to bf16 [N][K]
    prep_weights<<<8192, 256, 0, stream>>>(Wa_x, Wi_x, Wf_x, Wo_x,
                                           Wa_h, Wi_h, Wf_h, Wo_h, WxT, WhT);
    // 2) zero h0 (both parities) and s0 (every call: determinism)
    init_state<<<256, 256, 0, stream>>>(hbuf, sbuf);
    // 3) input projections for all timesteps (bf16 WMMA, f32 accumulate + bias)
    xproj_kernel<<<8192, 256, 0, stream>>>(X, WxT, ba, bi, bfp, bo, Xg);
    // 4) 512 recurrent steps, h ping-pongs between parity buffers
    for (int t = 0; t < T_STEPS; ++t) {
        const __bf16* hp = hbuf + (size_t)(t & 1) * BATCH * HDIM;
        __bf16*       hn = hbuf + (size_t)((t + 1) & 1) * BATCH * HDIM;
        lstm_step_kernel<<<128, 128, 0, stream>>>(hp, hn, WhT, Xg, sbuf,
                                                  (float*)d_out, t);
    }
}
